// Net_22840636080474
// MI455X (gfx1250) — matrix operand order repaired and checked by reading, not compiled, not run on hardware
//
#include <hip/hip_runtime.h>
#include <hip/hip_bf16.h>

typedef __attribute__((ext_vector_type(16))) __bf16 v16bf;
typedef __attribute__((ext_vector_type(8)))  float  v8f;

#define FIN  512
#define HDIM 16
#define CDIM 2

// ---------------------------------------------------------------------------
// gcn_norm pieces
// ---------------------------------------------------------------------------
__global__ void k_init_deg(float* __restrict__ deg, int n) {
    int i = blockIdx.x * blockDim.x + threadIdx.x;
    if (i < n) deg[i] = 1.0f;                      // self-loop weight
}

__global__ void k_acc_deg(const long long* __restrict__ col,
                          const float* __restrict__ ew,
                          float* __restrict__ deg, int e_cnt) {
    int e = blockIdx.x * blockDim.x + threadIdx.x;
    if (e < e_cnt) atomicAdd(&deg[(int)col[e]], ew[e]);
}

__global__ void k_dinv(const float* __restrict__ deg, float* __restrict__ dinv, int n) {
    int i = blockIdx.x * blockDim.x + threadIdx.x;
    if (i < n) {
        float d = deg[i];
        dinv[i] = d > 0.0f ? __frsqrt_rn(d) : 0.0f;
    }
}

// ---------------------------------------------------------------------------
// Layer-1 GEMM: hw[N,16] = x[N,512] @ W1[512,16] via v_wmma_f32_16x16x32_bf16.
// B (W1 as bf16, per-lane fragment layout) staged in LDS [kstep][lane][16].
// K-loop kept ROLLED so only one B fragment is live at a time (no LICM
// hoist -> no spills). Each wave does TWO row-tiles per pass: two independent
// WMMA chains share each B fragment (2x reuse, 2x ILP on the acc chain).
// ---------------------------------------------------------------------------
__global__ void k_gemm1_wmma(const float* __restrict__ x,
                             const float* __restrict__ W1,
                             float* __restrict__ hw, int n_tiles) {
    __shared__ __align__(16) __bf16 bsh[16 * 32 * 16];   // 8192 bf16 = 16 KB

    const int tid = threadIdx.x;
    // Cooperative fill: bsh[ks][lane][i] = bf16(W1[(ks*32 + 8*(lane>>4) + i + 8*(i>>3))*16 + (lane&15)])
    for (int idx = tid; idx < 16 * 32 * 16; idx += blockDim.x) {
        const int i    = idx & 15;
        const int lane = (idx >> 4) & 31;
        const int ks   = idx >> 9;
        const int k    = ks * 32 + ((lane >> 4) << 3) + i + ((i >> 3) << 3);   // lanes 0-15: K=0..7,16..23; 16-31: K=8..15,24..31
        bsh[idx] = (__bf16)W1[k * HDIM + (lane & 15)];       // column = lane&15
    }
    __syncthreads();

    const int lane    = tid & 31;
    const int wave    = (blockIdx.x * blockDim.x + tid) >> 5;
    const int nwaves  = (gridDim.x * blockDim.x) >> 5;
    const int nsel    = lane & 15;
    const int hi      = lane >> 4;        // 0: lower half-lanes, 1: upper
    const int a_khalf = hi * 8;           // A: K offset {0 | 8} within a 32-step
    const int npairs  = (n_tiles + 1) >> 1;

    for (int pair = wave; pair < npairs; pair += nwaves) {
        const int  t0   = pair * 2;
        const int  t1   = t0 + 1;
        const bool has1 = (t1 < n_tiles);                     // wave-uniform
        const float* xr0 = x + (size_t)(t0 * 16 + nsel) * FIN;
        const float* xr1 = x + (size_t)((has1 ? t1 : t0) * 16 + nsel) * FIN;

        v8f acc0 = {}, acc1 = {};
#pragma unroll 1
        for (int ks = 0; ks < 16; ++ks) {
            const int k0 = ks * 32 + a_khalf;
            v16bf a0, a1;
#pragma unroll
            for (int i = 0; i < 8; ++i) {
                a0[i]     = (__bf16)xr0[k0 + i];          // K = k0+0..7
                a0[8 + i] = (__bf16)xr0[k0 + 16 + i];     // K = k0+16..23
            }
#pragma unroll
            for (int i = 0; i < 8; ++i) {
                a1[i]     = (__bf16)xr1[k0 + i];
                a1[8 + i] = (__bf16)xr1[k0 + 16 + i];
            }
            const v16bf b = *(const v16bf*)&bsh[(ks * 32 + lane) * 16];   // 32B aligned
            acc0 = __builtin_amdgcn_wmma_f32_16x16x32_bf16(
                false, a0, false, b, (short)0, acc0, false, false);
            acc1 = __builtin_amdgcn_wmma_f32_16x16x32_bf16(
                false, a1, false, b, (short)0, acc1, false, false);
        }
        // C/D: VGPR j -> M = j + 8*hi, N = nsel
        const int mb0 = t0 * 16 + 8 * hi;
#pragma unroll
        for (int j = 0; j < 8; ++j)
            hw[(size_t)(mb0 + j) * HDIM + nsel] = acc0[j];
        if (has1) {
            const int mb1 = t1 * 16 + 8 * hi;
#pragma unroll
            for (int j = 0; j < 8; ++j)
                hw[(size_t)(mb1 + j) * HDIM + nsel] = acc1[j];
        }
    }
}

// ---------------------------------------------------------------------------
// Propagation layer 1: out1 initialized with self-loop term, then edge scatter
// ---------------------------------------------------------------------------
__global__ void k_init_out1(const float* __restrict__ dinv,
                            const float* __restrict__ hw,
                            float* __restrict__ out1, int n) {
    int idx = blockIdx.x * blockDim.x + threadIdx.x;   // n*16 threads
    if (idx >= n * HDIM) return;
    int i = idx >> 4;
    float di = dinv[i];
    out1[idx] = di * di * hw[idx];                     // norm_self = dinv^2 * 1
}

__global__ void k_prop1(const long long* __restrict__ row,
                        const long long* __restrict__ col,
                        const float* __restrict__ ew,
                        const float* __restrict__ dinv,
                        const float* __restrict__ hw,
                        float* __restrict__ out1, int e_cnt) {
    long long idx = (long long)blockIdx.x * blockDim.x + threadIdx.x;  // E*16 threads
    long long e = idx >> 4;
    int f = (int)(idx & 15);
    if (e >= e_cnt) return;
    int r = (int)row[e], c = (int)col[e];
    float nrm = dinv[r] * ew[e] * dinv[c];
    atomicAdd(&out1[(size_t)c * HDIM + f], nrm * hw[(size_t)r * HDIM + f]);
}

// ---------------------------------------------------------------------------
// Fused: h1 = relu(out1 + b1);  hw2[N,2] = h1 @ W2
// ---------------------------------------------------------------------------
__global__ void k_relu_gemm2(const float* __restrict__ out1,
                             const float* __restrict__ b1,
                             const float* __restrict__ W2,
                             float* __restrict__ hw2, int n) {
    int i = blockIdx.x * blockDim.x + threadIdx.x;
    if (i >= n) return;
    float o0 = 0.0f, o1 = 0.0f;
#pragma unroll
    for (int f = 0; f < HDIM; ++f) {
        float h = out1[(size_t)i * HDIM + f] + b1[f];
        h = h > 0.0f ? h : 0.0f;
        o0 += h * W2[f * CDIM + 0];
        o1 += h * W2[f * CDIM + 1];
    }
    hw2[(size_t)i * CDIM + 0] = o0;
    hw2[(size_t)i * CDIM + 1] = o1;
}

// ---------------------------------------------------------------------------
// Propagation layer 2 (C=2 per edge) + final bias/log_softmax
// ---------------------------------------------------------------------------
__global__ void k_init_out2(const float* __restrict__ dinv,
                            const float* __restrict__ hw2,
                            float* __restrict__ out2, int n) {
    int idx = blockIdx.x * blockDim.x + threadIdx.x;   // n*2 threads
    if (idx >= n * CDIM) return;
    int i = idx >> 1;
    float di = dinv[i];
    out2[idx] = di * di * hw2[idx];
}

__global__ void k_prop2(const long long* __restrict__ row,
                        const long long* __restrict__ col,
                        const float* __restrict__ ew,
                        const float* __restrict__ dinv,
                        const float* __restrict__ hw2,
                        float* __restrict__ out2, int e_cnt) {
    int e = blockIdx.x * blockDim.x + threadIdx.x;
    if (e >= e_cnt) return;
    int r = (int)row[e], c = (int)col[e];
    float nrm = dinv[r] * ew[e] * dinv[c];
    atomicAdd(&out2[(size_t)c * CDIM + 0], nrm * hw2[(size_t)r * CDIM + 0]);
    atomicAdd(&out2[(size_t)c * CDIM + 1], nrm * hw2[(size_t)r * CDIM + 1]);
}

__global__ void k_logsoftmax(const float* __restrict__ out2,
                             const float* __restrict__ b2,
                             float* __restrict__ out, int n) {
    int i = blockIdx.x * blockDim.x + threadIdx.x;
    if (i >= n) return;
    float v0 = out2[(size_t)i * CDIM + 0] + b2[0];
    float v1 = out2[(size_t)i * CDIM + 1] + b2[1];
    float mx = fmaxf(v0, v1);
    float lse = mx + __logf(__expf(v0 - mx) + __expf(v1 - mx));
    out[(size_t)i * CDIM + 0] = v0 - lse;
    out[(size_t)i * CDIM + 1] = v1 - lse;
}

// ---------------------------------------------------------------------------
// Host launcher
// ---------------------------------------------------------------------------
static inline size_t align256(size_t x) { return (x + 255) & ~(size_t)255; }

extern "C" void kernel_launch(void* const* d_in, const int* in_sizes, int n_in,
                              void* d_out, int out_size, void* d_ws, size_t ws_size,
                              hipStream_t stream) {
    const float*     x   = (const float*)d_in[0];
    const long long* ei  = (const long long*)d_in[1];   // int64 [2,E]
    const float*     ew  = (const float*)d_in[2];
    const float*     W1  = (const float*)d_in[3];
    const float*     b1  = (const float*)d_in[4];
    const float*     W2  = (const float*)d_in[5];
    const float*     b2  = (const float*)d_in[6];
    float*           out = (float*)d_out;

    const int n = in_sizes[0] / FIN;       // 100000
    const int e = in_sizes[2];             // 3200000
    const long long* row = ei;             // edge_index[0]
    const long long* col = ei + e;         // edge_index[1]

    // Workspace carve-up (256B aligned)
    char*  ws   = (char*)d_ws;
    size_t off  = 0;
    float* deg  = (float*)(ws + off); off = align256(off + (size_t)n * 4);
    float* dinv = (float*)(ws + off); off = align256(off + (size_t)n * 4);
    float* hw   = (float*)(ws + off); off = align256(off + (size_t)n * HDIM * 4);
    float* out1 = (float*)(ws + off); off = align256(off + (size_t)n * HDIM * 4);
    float* hw2  = (float*)(ws + off); off = align256(off + (size_t)n * CDIM * 4);
    float* out2 = (float*)(ws + off); off = align256(off + (size_t)n * CDIM * 4);
    (void)ws_size;

    const int T = 256;
    // 1) symmetric normalization
    k_init_deg<<<(n + T - 1) / T, T, 0, stream>>>(deg, n);
    k_acc_deg <<<(e + T - 1) / T, T, 0, stream>>>(col, ew, deg, e);
    k_dinv    <<<(n + T - 1) / T, T, 0, stream>>>(deg, dinv, n);

    // 2) WMMA GEMM: hw = x @ W1
    const int n_tiles = n / 16;            // 6250 (N divisible by 16)
    k_gemm1_wmma<<<256, 256, 0, stream>>>(x, W1, hw, n_tiles);

    // 3) layer-1 propagation (self-loop init + edge scatter)
    k_init_out1<<<(n * HDIM + T - 1) / T, T, 0, stream>>>(dinv, hw, out1, n);
    long long p1 = (long long)e * HDIM;
    k_prop1<<<(unsigned)((p1 + T - 1) / T), T, 0, stream>>>(row, col, ew, dinv, hw, out1, e);

    // 4) relu + bias + layer-2 GEMM
    k_relu_gemm2<<<(n + T - 1) / T, T, 0, stream>>>(out1, b1, W2, hw2, n);

    // 5) layer-2 propagation
    k_init_out2<<<(n * CDIM + T - 1) / T, T, 0, stream>>>(dinv, hw2, out2, n);
    k_prop2<<<(e + T - 1) / T, T, 0, stream>>>(row, col, ew, dinv, hw2, out2, e);

    // 6) bias + log_softmax
    k_logsoftmax<<<(n + T - 1) / T, T, 0, stream>>>(out2, b2, out, n);
}
